// RadarTpaLSTM_54357106098848
// MI455X (gfx1250) — matrix operand order repaired and checked by reading, not compiled, not run on hardware
//
#include <hip/hip_runtime.h>
#include <math.h>
#include <stdint.h>

// Problem constants (from reference)
#define B_    1024
#define T_    64
#define D_    512
#define HD_   512
#define L_    3
#define FN_   64
#define ALEN_ 63          // ATTN_LEN = T-1
#define G4_   2048        // 4*HD

typedef __attribute__((ext_vector_type(16))) __bf16 v16bf;
typedef __attribute__((ext_vector_type(8)))  float  v8f;
typedef __attribute__((ext_vector_type(4)))  int    i4;

struct __align__(16) U4 { unsigned x, y, z, w; };

__device__ __forceinline__ unsigned short f2bf(float x) {
  unsigned u = __float_as_uint(x);
  u += 0x7fffu + ((u >> 16) & 1u);          // round-to-nearest-even bf16
  return (unsigned short)(u >> 16);
}
__device__ __forceinline__ float sigf(float x) { return 1.f / (1.f + __expf(-x)); }

// ---------------------------------------------------------------------------
// CDNA5 async global->LDS copy (ASYNCcnt-tracked). Signature per this
// toolchain: (v4i AS1* global_src, v4i AS3* lds_dst, imm offset, imm cpol).
// Falls back to a direct 16B copy if the builtin is unavailable.
// ---------------------------------------------------------------------------
#if defined(__HIP_DEVICE_COMPILE__) && __has_builtin(__builtin_amdgcn_global_load_async_to_lds_b128)
#define HAVE_ASYNC_LDS 1
#else
#define HAVE_ASYNC_LDS 0
#endif

typedef __attribute__((address_space(1))) i4 gi4;
typedef __attribute__((address_space(3))) i4 li4;

__device__ __forceinline__ void async_cp_b128(const void* g, void* lds) {
#if HAVE_ASYNC_LDS
  __builtin_amdgcn_global_load_async_to_lds_b128(
      (gi4*)(uintptr_t)g,
      (li4*)(uint32_t)(uintptr_t)lds,
      0 /*offset*/, 0 /*cpol*/);
#else
  *(U4*)lds = *(const U4*)g;
#endif
}

template <int N>
__device__ __forceinline__ void wait_async() {
#if HAVE_ASYNC_LDS
#if __has_builtin(__builtin_amdgcn_s_wait_asynccnt)
  __builtin_amdgcn_s_wait_asynccnt(N);
#else
  asm volatile("s_wait_asynccnt %0" :: "i"(N));
#endif
#endif
}

// ---------------------------------------------------------------------------
// C[M,N] (f32, ld=N) = A[M,K](bf16, ld=lda) * W[N,K]^T(bf16) [+C_in if accum]
//                                                     [+bias0+bias1 if !accum]
// Block: 128x128 tile, 256 threads = 8 wave32 waves in a 4(M) x 2(N) grid.
// Each wave: 32x64 = 2x4 WMMA 16x16 f32 accumulators.
// K staged 32 at a time into double-buffered LDS by the async global->LDS
// engine; copy of tile k+1 overlaps WMMA of tile k.
// Out-of-range tile rows are CLAMPED (not branched) so EXEC stays all-1s for
// the async ops / WMMA; they only ever feed dead (unstored) output rows/cols.
// ---------------------------------------------------------------------------
__global__ __launch_bounds__(256)
void gemm_bf16_wmma(const unsigned short* __restrict__ A, int lda,
                    const unsigned short* __restrict__ W,
                    float* __restrict__ C,
                    const float* __restrict__ bias0,
                    const float* __restrict__ bias1,
                    int M, int N, int K, int accum)
{
  __shared__ __align__(16) unsigned short As[2][128 * 32];  // 2 x 8 KB
  __shared__ __align__(16) unsigned short Bs[2][128 * 32];  // 2 x 8 KB

  const int tid  = threadIdx.x;
  const int lane = tid & 31;
  const int wid  = tid >> 5;
  const int wm   = wid >> 1;            // 0..3
  const int wn   = wid & 1;             // 0..1
  const int row0 = blockIdx.y * 128;
  const int col0 = blockIdx.x * 128;
  const int lr   = lane & 15;
  const int lh   = lane >> 4;           // half-wave select (K interleave)

  // staging assignment: thread -> (tile row, 16-element K chunk)
  const int ldRow = tid >> 1;           // 0..127
  const int ldCol = (tid & 1) * 16;     // 0 or 16
  const int gr = min(row0 + ldRow, M - 1);   // clamp (dead rows only)
  const int gn = min(col0 + ldRow, N - 1);   // clamp (dead cols only)
  const unsigned short* gA = A + (long)gr * lda + ldCol;
  const unsigned short* gW = W + (long)gn * K + ldCol;
  const int ldsOff = ldRow * 32 + ldCol;

  v8f acc[2][4] = {};

  const int nk = K >> 5;                // K/32 (all K are multiples of 32)

  // prologue: stage tile 0 into buffer 0
  async_cp_b128(gA,     &As[0][ldsOff]);
  async_cp_b128(gA + 8, &As[0][ldsOff + 8]);
  async_cp_b128(gW,     &Bs[0][ldsOff]);
  async_cp_b128(gW + 8, &Bs[0][ldsOff + 8]);

  for (int ki = 0; ki < nk; ++ki) {
    const int buf = ki & 1;
    if (ki + 1 < nk) {
      // overlap: issue async copy of next K tile into the other buffer
      const int k1 = (ki + 1) << 5;
      async_cp_b128(gA + k1,     &As[buf ^ 1][ldsOff]);
      async_cp_b128(gA + k1 + 8, &As[buf ^ 1][ldsOff + 8]);
      async_cp_b128(gW + k1,     &Bs[buf ^ 1][ldsOff]);
      async_cp_b128(gW + k1 + 8, &Bs[buf ^ 1][ldsOff + 8]);
      wait_async<4>();                  // current tile's 4 copies done (in-order)
    } else {
      wait_async<0>();
    }
    __syncthreads();

    // ---- WMMA fragments per the CDNA5 16-bit VGPR layouts ----
    union FragA { U4 q[2]; v16bf v; };
    FragA fa[2];
    #pragma unroll
    for (int tm = 0; tm < 2; ++tm) {
      // A 16x32: lane lr = row; lanes 0-15 hold K {0..7,16..23}, 16-31 {8..15,24..31}
      const unsigned short* ap = &As[buf][(wm * 32 + tm * 16 + lr) * 32];
      fa[tm].q[0] = *(const U4*)(ap + lh * 8);
      fa[tm].q[1] = *(const U4*)(ap + 16 + lh * 8);
    }
    v16bf fb[4];
    #pragma unroll
    for (int tn = 0; tn < 4; ++tn) {
      // B 32x16: lane lr = col; lanes 0-15 hold K {0..7,16..23}, 16-31 {8..15,24..31}
      const unsigned short* bp = &Bs[buf][(wn * 64 + tn * 16 + lr) * 32];
      { FragA fbt; fbt.q[0] = *(const U4*)(bp + lh * 8); fbt.q[1] = *(const U4*)(bp + 16 + lh * 8); fb[tn] = fbt.v; }
    }

    #pragma unroll
    for (int tm = 0; tm < 2; ++tm)
      #pragma unroll
      for (int tn = 0; tn < 4; ++tn)
        acc[tm][tn] = __builtin_amdgcn_wmma_f32_16x16x32_bf16(
            false, fa[tm].v, false, fb[tn], (short)0, acc[tm][tn], false, false);

    __syncthreads();   // everyone done reading buf before it is refilled
  }

  // ---- epilogue: C/D layout -> VGPR r : M = r (lanes 0-15) / 8+r (16-31) ----
  #pragma unroll
  for (int tm = 0; tm < 2; ++tm)
    for (int tn = 0; tn < 4; ++tn)
      #pragma unroll
      for (int r = 0; r < 8; ++r) {
        const int row = row0 + wm * 32 + tm * 16 + ((lane < 16) ? r : 8 + r);
        const int col = col0 + wn * 64 + tn * 16 + lr;
        if (row < M && col < N) {
          float v = acc[tm][tn][r];
          const long idx = (long)row * N + col;
          if (accum) v += C[idx];
          else {
            if (bias0) v += bias0[col];
            if (bias1) v += bias1[col];
          }
          C[idx] = v;
        }
      }
}

// ---------------------------------------------------------------------------
// One-time fp32 -> bf16 conversion (weights / x converted once per launch,
// reused 64x by the recurrence -> conversion VALU leaves the hot loop).
// ---------------------------------------------------------------------------
__global__ void f32_to_bf16(const float* __restrict__ src,
                            unsigned short* __restrict__ dst, long n)
{
  const long i = (long)blockIdx.x * blockDim.x + threadIdx.x;
  if (i < n) dst[i] = f2bf(src[i]);
}

// ---------------------------------------------------------------------------
// Fused LSTM cell: gates [B,4H] f32 -> h (bf16, GEMM operand) , c (f32);
// optional f32 ys write for the attention head.
// ---------------------------------------------------------------------------
__global__ __launch_bounds__(256)
void lstm_cell(const float* __restrict__ gates,
               unsigned short* __restrict__ hbf,
               float* __restrict__ c,
               float* __restrict__ ys_t)
{
  const int idx = blockIdx.x * blockDim.x + threadIdx.x;   // B*HD threads
  const int b = idx >> 9, j = idx & 511;
  const float* g = gates + (long)b * G4_;
  const float i  = sigf(g[j]);
  const float f  = sigf(g[HD_ + j]);
  const float gg = tanhf(g[2 * HD_ + j]);
  const float o  = sigf(g[3 * HD_ + j]);
  const float cn = f * c[idx] + i * gg;
  const float hn = o * tanhf(cn);
  c[idx]   = cn;
  hbf[idx] = f2bf(hn);
  if (ys_t) ys_t[idx] = hn;
}

// ---------------------------------------------------------------------------
// Fused conv + TPA attention head, one workgroup per batch row.
// Stages relu(H[b]) (63x512 f32) + conv kernel in dynamic LDS (~145KB,
// within CDNA5's 320KB/WGP budget).
// ---------------------------------------------------------------------------
__global__ __launch_bounds__(256)
void tpa_head(const float* __restrict__ ys,     // [T,B,HD]
              const float* __restrict__ convW,  // [FN,1,63,1]
              const float* __restrict__ convb,  // [FN]
              const float* __restrict__ wbuf,   // [B,FN]  (lin1 output)
              float* __restrict__ vbuf)         // [B,FN]
{
  extern __shared__ float smem[];
  float* Hs   = smem;                   // 63*512
  float* kWs  = Hs + ALEN_ * HD_;       // 64*63
  float* wsh  = kWs + FN_ * ALEN_;      // 64
  float* vacc = wsh + FN_;              // 64
  const int b = blockIdx.x, tid = threadIdx.x;

  for (int i = tid; i < ALEN_ * HD_; i += 256) {
    const int t = i >> 9, f = i & 511;
    const float x = ys[(long)t * B_ * HD_ + (long)b * HD_ + f];
    Hs[i] = fmaxf(x, 0.f);
  }
  for (int i = tid; i < FN_ * ALEN_; i += 256) kWs[i] = convW[i];
  if (tid < FN_) { wsh[tid] = wbuf[b * FN_ + tid]; vacc[tid] = 0.f; }
  __syncthreads();

  for (int f = tid; f < HD_; f += 256) {
    float cv[FN_];
    #pragma unroll 1
    for (int n = 0; n < FN_; ++n) {
      // conv_vecs[b,f,n] = relu(conv_bcf[b,c,ff]) with c*512+ff = f*64+n (reshape)
      const int flat = f * FN_ + n;
      const int cc = flat >> 9, ff = flat & 511;
      float a = convb[cc];
      for (int t = 0; t < ALEN_; ++t) a += Hs[t * HD_ + ff] * kWs[cc * ALEN_ + t];
      cv[n] = fmaxf(a, 0.f);
    }
    float s = 0.f;
    for (int n = 0; n < FN_; ++n) s += cv[n] * wsh[n];
    const float alpha = sigf(s);
    for (int n = 0; n < FN_; ++n) atomicAdd(&vacc[n], alpha * cv[n]);
  }
  __syncthreads();
  if (tid < FN_) vbuf[b * FN_ + tid] = vacc[tid];
}

// concat [htt(bf16), v(f32->bf16)] -> catbf [B, 576] bf16 (lin2 GEMM operand)
__global__ void concat_bf16(const unsigned short* __restrict__ httbf,
                            const float* __restrict__ v,
                            unsigned short* __restrict__ cat)
{
  const int i = blockIdx.x * blockDim.x + threadIdx.x;
  if (i >= B_ * (HD_ + FN_)) return;
  const int b = i / (HD_ + FN_), j = i % (HD_ + FN_);
  cat[i] = (j < HD_) ? httbf[(long)b * HD_ + j] : f2bf(v[b * FN_ + (j - HD_)]);
}

__global__ void out_proj(const float* __restrict__ nh,
                         const float* __restrict__ oW,
                         const float* __restrict__ ob,
                         float* __restrict__ y)
{
  const int b = blockIdx.x * blockDim.x + threadIdx.x;
  if (b >= B_) return;
  float s = ob[0];
  for (int j = 0; j < HD_; ++j) s += nh[(long)b * HD_ + j] * oW[j];
  y[b] = s;
}

__global__ void zero_kernel(float* p, long n)
{
  const long i = (long)blockIdx.x * blockDim.x + threadIdx.x;
  if (i < n) p[i] = 0.f;
}

// ---------------------------------------------------------------------------
extern "C" void kernel_launch(void* const* d_in, const int* in_sizes, int n_in,
                              void* d_out, int out_size, void* d_ws, size_t ws_size,
                              hipStream_t stream)
{
  const float* x     = (const float*)d_in[0];
  const float* W_ih  = (const float*)d_in[1];
  const float* W_hh  = (const float*)d_in[2];
  const float* b_ih  = (const float*)d_in[3];
  const float* b_hh  = (const float*)d_in[4];
  const float* convW = (const float*)d_in[5];
  const float* convb = (const float*)d_in[6];
  const float* l1W   = (const float*)d_in[7];
  const float* l1b   = (const float*)d_in[8];
  const float* l2W   = (const float*)d_in[9];
  const float* l2b   = (const float*)d_in[10];
  const float* oW    = (const float*)d_in[11];
  const float* ob    = (const float*)d_in[12];
  float* out = (float*)d_out;

  // -------- workspace carve-out (~225 MB; all chunks 16B-aligned) --------
  char* ws = (char*)d_ws;
  float* gates = (float*)ws; ws += (size_t)B_ * G4_ * 4;            // 8 MB f32
  float* c     = (float*)ws; ws += (size_t)3 * B_ * HD_ * 4;        // 6 MB f32
  float* ys    = (float*)ws; ws += (size_t)T_ * B_ * HD_ * 4;       // 128 MB f32
  float* wbuf  = (float*)ws; ws += (size_t)B_ * FN_ * 4;
  float* vbuf  = (float*)ws; ws += (size_t)B_ * FN_ * 4;
  float* nh    = (float*)ws; ws += (size_t)B_ * HD_ * 4;
  unsigned short* xbf   = (unsigned short*)ws; ws += (size_t)B_ * T_ * D_ * 2;   // 64 MB
  unsigned short* Wihbf = (unsigned short*)ws; ws += (size_t)L_ * G4_ * HD_ * 2; // 6.3 MB
  unsigned short* Whhbf = (unsigned short*)ws; ws += (size_t)L_ * G4_ * HD_ * 2; // 6.3 MB
  unsigned short* hbf   = (unsigned short*)ws; ws += (size_t)3 * B_ * HD_ * 2;   // 3 MB
  unsigned short* catbf = (unsigned short*)ws; ws += (size_t)B_ * (HD_ + FN_) * 2;
  unsigned short* l1Wbf = (unsigned short*)ws; ws += (size_t)FN_ * HD_ * 2;
  unsigned short* l2Wbf = (unsigned short*)ws; ws += (size_t)HD_ * (HD_ + FN_) * 2;

  const dim3 blk(256);

  // -------- one-time precision conversion (weights L2-resident as bf16) ---
  {
    const long nx = (long)B_ * T_ * D_;
    f32_to_bf16<<<dim3((unsigned)((nx + 255) / 256)), blk, 0, stream>>>(x, xbf, nx);
    const long nw = (long)L_ * G4_ * HD_;
    f32_to_bf16<<<dim3((unsigned)((nw + 255) / 256)), blk, 0, stream>>>(W_ih, Wihbf, nw);
    f32_to_bf16<<<dim3((unsigned)((nw + 255) / 256)), blk, 0, stream>>>(W_hh, Whhbf, nw);
    const long n1 = (long)FN_ * HD_;
    f32_to_bf16<<<dim3((unsigned)((n1 + 255) / 256)), blk, 0, stream>>>(l1W, l1Wbf, n1);
    const long n2 = (long)HD_ * (HD_ + FN_);
    f32_to_bf16<<<dim3((unsigned)((n2 + 255) / 256)), blk, 0, stream>>>(l2W, l2Wbf, n2);
  }
  // zero-init recurrent state: c (f32) and hbf (bf16, zero bits == 0.0bf16)
  {
    const long nc = (long)3 * B_ * HD_;
    zero_kernel<<<dim3((unsigned)((nc + 255) / 256)), blk, 0, stream>>>(c, nc);
    const long nh2 = (long)3 * B_ * HD_ / 2;     // ushorts viewed as f32 pairs
    zero_kernel<<<dim3((unsigned)((nh2 + 255) / 256)), blk, 0, stream>>>((float*)hbf, nh2);
  }

  const dim3 gGate(G4_ / 128, B_ / 128);   // 16 x 8 blocks

  for (int t = 0; t < T_; ++t) {
    for (int l = 0; l < L_; ++l) {
      const unsigned short* Ain =
          (l == 0) ? (xbf + (size_t)t * D_) : (hbf + (size_t)(l - 1) * B_ * HD_);
      const int lda = (l == 0) ? T_ * D_ : HD_;
      // gates = Ain @ W_ih[l]^T + b_ih[l] + b_hh[l]
      gemm_bf16_wmma<<<gGate, blk, 0, stream>>>(
          Ain, lda, Wihbf + (size_t)l * G4_ * HD_, gates,
          b_ih + (size_t)l * G4_, b_hh + (size_t)l * G4_, B_, G4_, HD_, 0);
      // gates += h[l] @ W_hh[l]^T
      gemm_bf16_wmma<<<gGate, blk, 0, stream>>>(
          hbf + (size_t)l * B_ * HD_, HD_, Whhbf + (size_t)l * G4_ * HD_, gates,
          nullptr, nullptr, B_, G4_, HD_, 1);
      float* yst = (l == L_ - 1) ? (ys + (size_t)t * B_ * HD_) : nullptr;
      lstm_cell<<<dim3((B_ * HD_) / 256), blk, 0, stream>>>(
          gates, hbf + (size_t)l * B_ * HD_, c + (size_t)l * B_ * HD_, yst);
    }
  }

  const float* htt = ys + (size_t)(T_ - 1) * B_ * HD_;      // f32 view (head)
  const unsigned short* httbf = hbf + (size_t)2 * B_ * HD_; // bf16 view (GEMMs)

  // w = htt @ lin1^T + lin1_b    (N=64, bounds-guarded/clamped in the gemm)
  gemm_bf16_wmma<<<dim3(1, B_ / 128), blk, 0, stream>>>(
      httbf, HD_, l1Wbf, wbuf, l1b, nullptr, B_, FN_, HD_, 0);

  // fused conv + attention (dynamic LDS ~145 KB; CDNA5 WGP has 320 KB)
  const size_t smemB = (size_t)(ALEN_ * HD_ + FN_ * ALEN_ + FN_ + FN_) * sizeof(float);
  tpa_head<<<dim3(B_), blk, smemB, stream>>>(ys, convW, convb, wbuf, vbuf);

  concat_bf16<<<dim3((B_ * (HD_ + FN_) + 255) / 256), blk, 0, stream>>>(httbf, vbuf, catbf);

  // new_ht = cat @ lin2^T + lin2_b   (K = 576)
  gemm_bf16_wmma<<<dim3(HD_ / 128, B_ / 128), blk, 0, stream>>>(
      catbf, HD_ + FN_, l2Wbf, nh, l2b, nullptr, B_, HD_, HD_ + FN_, 0);

  out_proj<<<dim3((B_ + 255) / 256), blk, 0, stream>>>(nh, oW, ob, out);
}
